// MultiHeadAttention_2276332667217
// MI455X (gfx1250) — compile-verified, matrix operand order checked by reading, not run on hardware
//
#include <hip/hip_runtime.h>
#include <hip/hip_bf16.h>
#include <stdint.h>

typedef _Float16 half_t;
typedef __attribute__((ext_vector_type(8)))  _Float16 h8;
typedef __attribute__((ext_vector_type(16))) _Float16 v16h;
typedef __attribute__((ext_vector_type(8)))  float    v8f;

#define D_MODEL 1024
#define N_HEAD  16
#define D_K     64
#define BATCH   4
#define SEQ     1024
#define ROWS    (BATCH * SEQ)   // 4096

// ---------------------------------------------------------------------------
// WMMA helper: D = A(16x32 f16) * B(32x16 f16) + C(16x16 f32)
// ---------------------------------------------------------------------------
__device__ inline v8f wmma_f16(v16h a, v16h b, v8f c) {
    return __builtin_amdgcn_wmma_f32_16x16x32_f16(
        /*neg_a=*/false, a, /*neg_b=*/false, b,
        /*c_mod=*/(short)0, c, /*reuse_a=*/false, /*reuse_b=*/false);
}

// Per-lane fragment load: 16 f16 K-values. Lane = 16*hi + lo.
// Chunks [k0+8*hi,+8) and [k0+16+8*hi,+8); 16B aligned -> b128 loads.
__device__ inline v16h load_frag16(const half_t* p, int hi) {
    union { struct { h8 a; h8 b; } s; v16h v; } u;
    u.s.a = *reinterpret_cast<const h8*>(p + 8 * hi);
    u.s.b = *reinterpret_cast<const h8*>(p + 16 + 8 * hi);
    return u.v;
}

// ---------------------------------------------------------------------------
// Async global->LDS copy of one contiguous 2KB tile (16x64 f16), one wave.
// Builtin signature (probe-confirmed via diagnostic): 4 args,
//   (int4 addrspace(1)*, int4 addrspace(3)*, imm offset, imm cpol)
// ---------------------------------------------------------------------------
#if __has_builtin(__builtin_amdgcn_global_load_async_to_lds_b128) && \
    __has_builtin(__builtin_amdgcn_s_wait_asynccnt)
#define ASYNC_MODE 2
#else
#define ASYNC_MODE 1   // inline asm per CDNA5 ISA
#endif

typedef __attribute__((ext_vector_type(4))) int i32x4;
typedef __attribute__((address_space(1))) i32x4 gvec_t;   // global b128 beat
typedef __attribute__((address_space(3))) i32x4 lvec_t;   // LDS b128 beat

__device__ inline void async_tile_2k(const half_t* gsrc, half_t* ldst, int lane) {
#if ASYNC_MODE == 2
    uint64_t g = (uint64_t)(uintptr_t)gsrc + (uint64_t)lane * 16;
    uint32_t l = (uint32_t)(uintptr_t)ldst + (uint32_t)lane * 16; // low 32b == LDS addr
    #pragma unroll
    for (int i = 0; i < 4; ++i)
        __builtin_amdgcn_global_load_async_to_lds_b128(
            (gvec_t*)(uintptr_t)(g + (uint64_t)(i * 512)),
            (lvec_t*)(uint32_t)(l + (uint32_t)(i * 512)), 0, 0);
#elif ASYNC_MODE == 1
    uint64_t g = (uint64_t)(uintptr_t)gsrc + (uint64_t)lane * 16;
    uint32_t l = (uint32_t)(uintptr_t)ldst + (uint32_t)lane * 16;
    #pragma unroll
    for (int i = 0; i < 4; ++i)
        asm volatile("global_load_async_to_lds_b128 %0, %1, off"
                     :: "v"(l + (uint32_t)(i * 512)), "v"(g + (uint64_t)(i * 512))
                     : "memory");
#else
    #pragma unroll
    for (int i = 0; i < 4; ++i)
        *reinterpret_cast<h8*>(ldst + lane * 8 + i * 256) =
            *reinterpret_cast<const h8*>(gsrc + lane * 8 + i * 256);
#endif
}

__device__ inline void async_wait_le4() {
#if ASYNC_MODE == 2
    __builtin_amdgcn_s_wait_asynccnt(4);
#elif ASYNC_MODE == 1
    asm volatile("s_wait_asynccnt 4" ::: "memory");
#endif
}
__device__ inline void async_wait_le0() {
#if ASYNC_MODE == 2
    __builtin_amdgcn_s_wait_asynccnt(0);
#elif ASYNC_MODE == 1
    asm volatile("s_wait_asynccnt 0" ::: "memory");
#endif
}

// ---------------------------------------------------------------------------
// 1) LayerNorm (f32 in) -> f16 out. One block per row of 1024.
// ---------------------------------------------------------------------------
__global__ __launch_bounds__(256)
void ln_kernel(const float* __restrict__ X, const float* __restrict__ gamma,
               const float* __restrict__ beta, half_t* __restrict__ out) {
    const int row = blockIdx.x;
    const int tid = threadIdx.x;
    const float* x = X + (size_t)row * D_MODEL;

    float s = 0.f, ss = 0.f;
    for (int i = tid; i < D_MODEL; i += 256) {
        float v = x[i];
        s += v; ss += v * v;
    }
    __shared__ float rs[256], rss[256];
    rs[tid] = s; rss[tid] = ss;
    __syncthreads();
    for (int off = 128; off > 0; off >>= 1) {
        if (tid < off) { rs[tid] += rs[tid + off]; rss[tid] += rss[tid + off]; }
        __syncthreads();
    }
    const float mu   = rs[0] * (1.0f / D_MODEL);
    const float var  = rss[0] * (1.0f / D_MODEL) - mu * mu;
    const float rinv = rsqrtf(var + 1e-6f);

    half_t* o = out + (size_t)row * D_MODEL;
    for (int i = tid; i < D_MODEL; i += 256)
        o[i] = (half_t)((x[i] - mu) * rinv * gamma[i] + beta[i]);
}

// ---------------------------------------------------------------------------
// 2) Weight cast + transpose: W[k][n] f32 -> WT[n][k] f16   (1024x1024)
// ---------------------------------------------------------------------------
__global__ __launch_bounds__(256)
void wcast_kernel(const float* __restrict__ W, half_t* __restrict__ WT) {
    int e = blockIdx.x * 256 + threadIdx.x;
    int n = e >> 10, k = e & 1023;
    WT[(size_t)n * 1024 + k] = (half_t)W[(size_t)k * 1024 + n];
}

// ---------------------------------------------------------------------------
// Register-blocked WMMA mainloop: 64x64 tile per wave, K=1024.
// 16 b128 loads : 16 wmma per k-step (1:1), acc = 4x4 v8f.
// ---------------------------------------------------------------------------
__device__ inline void gemm64_mainloop(const half_t* __restrict__ X,
                                       const half_t* __restrict__ WT,
                                       int m0, int n0, int lo, int hi,
                                       v8f acc[4][4]) {
    const half_t* xr0 = X  + (size_t)(m0 + lo) * D_MODEL;
    const half_t* wr0 = WT + (size_t)(n0 + lo) * D_MODEL;
    for (int k0 = 0; k0 < D_MODEL; k0 += 32) {
        v16h a[4], b[4];
        #pragma unroll
        for (int i = 0; i < 4; ++i)
            a[i] = load_frag16(xr0 + (size_t)i * 16 * D_MODEL + k0, hi);
        #pragma unroll
        for (int j = 0; j < 4; ++j)
            b[j] = load_frag16(wr0 + (size_t)j * 16 * D_MODEL + k0, hi);
        __builtin_prefetch(xr0 + k0 + 256, 0, 1);   // -> global_prefetch_b8
        __builtin_prefetch(wr0 + k0 + 256, 0, 1);
        #pragma unroll
        for (int i = 0; i < 4; ++i)
            #pragma unroll
            for (int j = 0; j < 4; ++j)
                acc[i][j] = wmma_f16(a[i], b[j], acc[i][j]);
    }
}

// ---------------------------------------------------------------------------
// 3) Projection GEMM: Y = X[4096,1024] * W + bias, scaled, f16 out,
//    head-major [b,h,s,d] or d-major [b,h,d,s] (for V).
// ---------------------------------------------------------------------------
__global__ __launch_bounds__(256)
void gemm_proj_wmma(const half_t* __restrict__ X, const half_t* __restrict__ WT,
                    const float* __restrict__ bias, half_t* __restrict__ Y,
                    float scale, int vtrans) {
    const int wave = threadIdx.x >> 5, lane = threadIdx.x & 31;
    const int lo = lane & 15, hi = lane >> 4;
    const int tile = blockIdx.x * 8 + wave;     // 1024 tiles of 64x64
    const int tn = tile & 15, tm = tile >> 4;   // N/64=16, M/64=64

    v8f acc[4][4] = {};
    gemm64_mainloop(X, WT, tm * 64, tn * 64, lo, hi, acc);

    #pragma unroll
    for (int j = 0; j < 4; ++j) {
        const int n = tn * 64 + j * 16 + lo;
        const int h = n >> 6, d = n & 63;
        const float bn = bias[n];
        #pragma unroll
        for (int i = 0; i < 4; ++i) {
            #pragma unroll
            for (int r = 0; r < 8; ++r) {
                const int m = tm * 64 + i * 16 + r + 8 * hi;
                const int bb = m >> 10, s = m & 1023;
                const float y = (acc[i][j][r] + bn) * scale;
                size_t idx = vtrans
                    ? ((size_t)(bb * N_HEAD + h) * D_K + d) * SEQ + s
                    : ((size_t)(bb * N_HEAD + h) * SEQ + s) * D_K + d;
                Y[idx] = (half_t)y;
            }
        }
    }
}

// ---------------------------------------------------------------------------
// 4) Attention: one wave per (b, h, 16-query tile).
//    Double-buffered async K tiles in LDS; scores in 64KB LDS;
//    masked softmax; attn -> d_out; P @ V -> o_tmp.
// ---------------------------------------------------------------------------
__global__ __launch_bounds__(32)
void attn_wmma(const half_t* __restrict__ qh,   // [b,h,s,d] (pre-scaled 1/8)
               const half_t* __restrict__ kh,   // [b,h,s,d]
               const half_t* __restrict__ vt,   // [b,h,d,s]
               const int* __restrict__ mask,    // [b,sq,sk]
               float* __restrict__ attn_out,    // [b,h,sq,sk]
               half_t* __restrict__ o_tmp) {    // [4096, 1024] f16
    __shared__ float sS[16 * SEQ];                       // 64 KB scores
    __shared__ __align__(16) half_t kbuf[2][16 * D_K];   // 2 x 2KB K tiles

    const int lane = threadIdx.x & 31;
    const int lo = lane & 15, hi = lane >> 4;
    const int blk = blockIdx.x;                 // b*16*64 + h*64 + qt
    const int qt = blk & 63;
    const int h  = (blk >> 6) & 15;
    const int b  = blk >> 10;

    const half_t* qrow = qh + ((size_t)(b * N_HEAD + h) * SEQ + qt * 16 + lo) * D_K;
    const v16h aq0 = load_frag16(qrow + 0,  hi);
    const v16h aq1 = load_frag16(qrow + 32, hi);

    // ---- scores: S[16, 1024] = (q/8) . k^T, async K-tile pipeline ----
    const half_t* kbase = kh + (size_t)(b * N_HEAD + h) * SEQ * D_K;
    async_tile_2k(kbase, &kbuf[0][0], lane);
    for (int kt = 0; kt < 64; ++kt) {
        if (kt < 63) {
            async_tile_2k(kbase + (size_t)(kt + 1) * 16 * D_K,
                          &kbuf[(kt + 1) & 1][0], lane);
            async_wait_le4();    // previous tile landed, next in flight
        } else {
            async_wait_le0();
        }
        const half_t* krow = &kbuf[kt & 1][lo * D_K];
        v16h bk0 = load_frag16(krow + 0,  hi);
        v16h bk1 = load_frag16(krow + 32, hi);
        v8f acc = {};
        acc = wmma_f16(aq0, bk0, acc);
        acc = wmma_f16(aq1, bk1, acc);
        const int col = kt * 16 + lo;
        #pragma unroll
        for (int r = 0; r < 8; ++r)
            sS[(r + 8 * hi) * SEQ + col] = acc[r];
    }
    __syncthreads();

    // ---- masked softmax: row = lo, lane-half hi scans 512 cols ----
    const int qglob = qt * 16 + lo;
    const int* mrow = mask + ((size_t)b * SEQ + qglob) * SEQ;
    float* srow = sS + lo * SEQ;
    const int c0 = hi * 512, c1 = c0 + 512;

    float mx = -3.0e38f;
    for (int c = c0; c < c1; ++c) {
        float s = srow[c];
        if (mrow[c] == 0) s = -1e9f;
        srow[c] = s;
        mx = fmaxf(mx, s);
    }
    mx = fmaxf(mx, __shfl_xor(mx, 16, 32));

    float sum = 0.f;
    for (int c = c0; c < c1; ++c) {
        float e = __expf(srow[c] - mx);
        srow[c] = e;
        sum += e;
    }
    sum += __shfl_xor(sum, 16, 32);
    const float inv = 1.0f / sum;

    float* arow = attn_out + ((size_t)(b * N_HEAD + h) * SEQ + qglob) * SEQ;
    for (int c = c0; c < c1; ++c) {
        float p = srow[c] * inv;
        srow[c] = p;
        arow[c] = p;
    }
    __syncthreads();

    // ---- out tile [16, 64] = P[16,1024] @ V[1024,64] ----
    const half_t* vbase = vt + (size_t)(b * N_HEAD + h) * D_K * SEQ;
    for (int nt = 0; nt < 4; ++nt) {
        const half_t* vrow = vbase + (size_t)(nt * 16 + lo) * SEQ;
        v8f acc = {};
        for (int k0 = 0; k0 < SEQ; k0 += 32) {
            const float* prow = sS + lo * SEQ + k0;
            v16h pa;
            #pragma unroll
            for (int j = 0; j < 8; ++j) pa[j]     = (half_t)prow[8 * hi + j];
            #pragma unroll
            for (int j = 0; j < 8; ++j) pa[8 + j] = (half_t)prow[16 + 8 * hi + j];
            v16h bv = load_frag16(vrow + k0, hi);
            acc = wmma_f16(pa, bv, acc);
        }
        #pragma unroll
        for (int r = 0; r < 8; ++r) {
            const int m   = b * SEQ + qt * 16 + r + 8 * hi;
            const int col = h * D_K + nt * 16 + lo;
            o_tmp[(size_t)m * D_MODEL + col] = (half_t)acc[r];
        }
    }
}

// ---------------------------------------------------------------------------
// 5) Output projection + bias + residual -> f32 d_out
// ---------------------------------------------------------------------------
__global__ __launch_bounds__(256)
void gemm_out_wmma(const half_t* __restrict__ X, const half_t* __restrict__ WT,
                   const float* __restrict__ bias, const float* __restrict__ resid,
                   float* __restrict__ Y) {
    const int wave = threadIdx.x >> 5, lane = threadIdx.x & 31;
    const int lo = lane & 15, hi = lane >> 4;
    const int tile = blockIdx.x * 8 + wave;
    const int tn = tile & 15, tm = tile >> 4;

    v8f acc[4][4] = {};
    gemm64_mainloop(X, WT, tm * 64, tn * 64, lo, hi, acc);

    #pragma unroll
    for (int j = 0; j < 4; ++j) {
        const int n = tn * 64 + j * 16 + lo;
        const float bn = bias[n];
        #pragma unroll
        for (int i = 0; i < 4; ++i) {
            #pragma unroll
            for (int r = 0; r < 8; ++r) {
                const int m = tm * 64 + i * 16 + r + 8 * hi;
                const size_t idx = (size_t)m * D_MODEL + n;
                Y[idx] = acc[i][j][r] + bn + resid[idx];
            }
        }
    }
}

// ---------------------------------------------------------------------------
// Host launcher
// ---------------------------------------------------------------------------
extern "C" void kernel_launch(void* const* d_in, const int* in_sizes, int n_in,
                              void* d_out, int out_size, void* d_ws, size_t ws_size,
                              hipStream_t stream) {
    const float* Q    = (const float*)d_in[0];
    const float* K    = (const float*)d_in[1];
    const float* V    = (const float*)d_in[2];
    const int*   mask = (const int*)  d_in[3];
    const float* lnqg = (const float*)d_in[4];
    const float* lnqb = (const float*)d_in[5];
    const float* lnkg = (const float*)d_in[6];
    const float* lnkb = (const float*)d_in[7];
    const float* lnvg = (const float*)d_in[8];
    const float* lnvb = (const float*)d_in[9];
    const float* wq   = (const float*)d_in[10];
    const float* bq   = (const float*)d_in[11];
    const float* wk   = (const float*)d_in[12];
    const float* bk   = (const float*)d_in[13];
    const float* wv   = (const float*)d_in[14];
    const float* bv   = (const float*)d_in[15];
    const float* wo   = (const float*)d_in[16];
    const float* bo   = (const float*)d_in[17];

    float* out_p  = (float*)d_out;                              // [4,1024,1024]
    float* attn_p = out_p + (size_t)BATCH * SEQ * D_MODEL;      // [4,16,1024,1024]

    char* w = (char*)d_ws;
    const size_t MB = 1024 * 1024;
    half_t* xq   = (half_t*)(w + 0 * MB);    // 8 MB, reused as o_tmp
    half_t* xk   = (half_t*)(w + 8 * MB);
    half_t* xv   = (half_t*)(w + 16 * MB);
    half_t* wqT  = (half_t*)(w + 24 * MB);
    half_t* wkT  = (half_t*)(w + 26 * MB);
    half_t* wvT  = (half_t*)(w + 28 * MB);
    half_t* woT  = (half_t*)(w + 30 * MB);
    half_t* q_h  = (half_t*)(w + 32 * MB);   // [b,h,s,d]
    half_t* k_h  = (half_t*)(w + 40 * MB);   // [b,h,s,d]
    half_t* v_t  = (half_t*)(w + 48 * MB);   // [b,h,d,s]
    half_t* o_tmp = xq;

    ln_kernel<<<ROWS, 256, 0, stream>>>(Q, lnqg, lnqb, xq);
    ln_kernel<<<ROWS, 256, 0, stream>>>(K, lnkg, lnkb, xk);
    ln_kernel<<<ROWS, 256, 0, stream>>>(V, lnvg, lnvb, xv);

    wcast_kernel<<<4096, 256, 0, stream>>>(wq, wqT);
    wcast_kernel<<<4096, 256, 0, stream>>>(wk, wkT);
    wcast_kernel<<<4096, 256, 0, stream>>>(wv, wvT);
    wcast_kernel<<<4096, 256, 0, stream>>>(wo, woT);

    // 1024 tiles of 64x64 / 8 waves per block = 128 blocks
    gemm_proj_wmma<<<128, 256, 0, stream>>>(xq, wqT, bq, q_h, 0.125f, 0);
    gemm_proj_wmma<<<128, 256, 0, stream>>>(xk, wkT, bk, k_h, 1.0f,   0);
    gemm_proj_wmma<<<128, 256, 0, stream>>>(xv, wvT, bv, v_t, 1.0f,   1);

    attn_wmma<<<4096, 32, 0, stream>>>(q_h, k_h, v_t, mask, attn_p, o_tmp);

    gemm_out_wmma<<<128, 256, 0, stream>>>(o_tmp, woT, bo, Q, out_p);
}
